// BertEmbeddingsWithVideo_10565619548629
// MI455X (gfx1250) — compile-verified, matrix operand order checked by reading, not run on hardware
//
#include <hip/hip_runtime.h>
#include <hip/hip_bf16.h>
#include <math.h>

// ---------------------------------------------------------------------------
// BertEmbeddingsWithVideo on MI455X (gfx1250, wave32, WMMA bf16)
// N=32, L=1024, H=768, DW=300(pad 320), DV=3072, DR=2048, T=2, V=30522
// ---------------------------------------------------------------------------

#define NTOK   32768        // N*L
#define HDIM   768
#define EPS_LN 1e-12f

typedef __attribute__((ext_vector_type(16))) __bf16 v16bf;
typedef __attribute__((ext_vector_type(8)))  float  v8f;

union FragB16 {
    v16bf v;
    struct { uint4 lo; uint4 hi; } q;
};

__device__ __forceinline__ unsigned short f2bf(float f) {
    union { float f; unsigned u; } v; v.f = f;
    unsigned u = v.u;
    unsigned r = u + 0x7FFFu + ((u >> 16) & 1u);   // round-to-nearest-even
    return (unsigned short)(r >> 16);
}

// ---------------------------------------------------------------------------
// Stage 1: (optional gather) -> LayerNorm over din -> bf16 (zero-padded to kpad)
// one block (256 threads) per token row
// ---------------------------------------------------------------------------
__global__ __launch_bounds__(256)
void ln1_to_bf16(const float* __restrict__ dense,      // (NTOK, din) or null
                 const int*   __restrict__ ids,        // (NTOK) or null
                 const float* __restrict__ table,      // (V, din) or null
                 const float* __restrict__ w,
                 const float* __restrict__ b,
                 unsigned short* __restrict__ A,       // (NTOK, kpad) bf16
                 int din, int kpad)
{
    const int t   = blockIdx.x;
    const int tid = threadIdx.x;
    const float* src = ids ? (table + (size_t)ids[t] * din)
                           : (dense + (size_t)t * din);

    float p1 = 0.f, p2 = 0.f;
    for (int d = tid; d < din; d += 256) {
        float x = src[d];
        p1 += x; p2 += x * x;
    }
    __shared__ float s1[256], s2[256];
    s1[tid] = p1; s2[tid] = p2;
    __syncthreads();
    for (int off = 128; off > 0; off >>= 1) {
        if (tid < off) { s1[tid] += s1[tid + off]; s2[tid] += s2[tid + off]; }
        __syncthreads();
    }
    const float inv  = 1.0f / (float)din;
    const float mean = s1[0] * inv;
    const float var  = s2[0] * inv - mean * mean;
    const float rstd = rsqrtf(var + EPS_LN);

    unsigned short* dst = A + (size_t)t * kpad;
    for (int d = tid; d < kpad; d += 256) {
        float v = 0.f;
        if (d < din) v = w[d] * (src[d] - mean) * rstd + b[d];
        dst[d] = f2bf(v);
    }
}

// ---------------------------------------------------------------------------
// Weight convert: (HDIM x din) f32 -> (HDIM x kpad) bf16, zero padded
// ---------------------------------------------------------------------------
__global__ __launch_bounds__(256)
void conv_w_bf16(const float* __restrict__ W, unsigned short* __restrict__ Wb,
                 int din, int kpad)
{
    const int total = HDIM * kpad;
    for (int i = blockIdx.x * 256 + threadIdx.x; i < total; i += gridDim.x * 256) {
        int h = i / kpad, k = i - h * kpad;
        float v = (k < din) ? W[(size_t)h * din + k] : 0.f;
        Wb[i] = f2bf(v);
    }
}

// ---------------------------------------------------------------------------
// GEMM (bf16 WMMA) + bias + ReLU + LayerNorm(H) fused, accumulate into acc.
// Block: 256 threads = 8 waves; 16 output rows x 768 cols per block.
// Wave w owns columns [w*96, w*96+96) = 6 WMMA N-tiles.
// ---------------------------------------------------------------------------
#define LNS (HDIM + 8)

template <int K>
__global__ __launch_bounds__(256)
void gemm_bias_relu_ln(const unsigned short* __restrict__ A,    // (NTOK, K) bf16
                       const unsigned short* __restrict__ Wb,   // (HDIM, K) bf16
                       const float* __restrict__ bias,          // (HDIM)
                       const float* __restrict__ lnw,
                       const float* __restrict__ lnb,
                       float* __restrict__ acc,                 // (NTOK, HDIM)
                       int beta)                                // 0: write, 1: add
{
    const int wv      = threadIdx.x >> 5;       // wave 0..7
    const int lane    = threadIdx.x & 31;
    const int r       = lane & 15;
    const int khalf   = (lane >> 4) << 3;       // 0 or 8 (bf16 A/B VGPR striping)
    const int rowBase = blockIdx.x << 4;        // 16 rows per block
    const int colBase = wv * 96;

    v8f c[6];
    #pragma unroll
    for (int j = 0; j < 6; ++j) c[j] = (v8f)0.f;

    const unsigned short* aRow = A + (size_t)(rowBase + r) * K + khalf;
    const unsigned short* bCol = Wb + (size_t)(colBase + r) * K + khalf;

    for (int k0 = 0; k0 < K; k0 += 32) {
        FragB16 a;
        a.q.lo = *(const uint4*)(aRow + k0);
        a.q.hi = *(const uint4*)(aRow + k0 + 16);
        if (k0 + 32 < K)  // pull next A panel lines toward L2/WGP$
            __builtin_prefetch(aRow + k0 + 32, 0, 1);

        FragB16 bf[6];
        #pragma unroll
        for (int j = 0; j < 6; ++j) {
            const unsigned short* bp = bCol + (size_t)(j * 16) * K + k0;
            bf[j].q.lo = *(const uint4*)(bp);
            bf[j].q.hi = *(const uint4*)(bp + 16);
        }
        #pragma unroll
        for (int j = 0; j < 6; ++j) {
            c[j] = __builtin_amdgcn_wmma_f32_16x16x32_bf16(
                       false, a.v, false, bf[j].v, (short)0, c[j], false, false);
        }
    }

    // bias + ReLU into LDS tile (16 x 768)
    __shared__ float lds[16 * LNS];
    #pragma unroll
    for (int j = 0; j < 6; ++j) {
        const int col = colBase + j * 16 + r;
        const float bcol = bias[col];
        #pragma unroll
        for (int i = 0; i < 8; ++i) {
            const int row = i + khalf;          // khalf doubles as the M offset (0/8)
            lds[row * LNS + col] = fmaxf(c[j][i] + bcol, 0.f);
        }
    }
    __syncthreads();

    // LayerNorm over H=768: wave wv handles rows {wv, wv+8}
    #pragma unroll
    for (int rr = 0; rr < 2; ++rr) {
        const int row = wv + rr * 8;
        float p1 = 0.f, p2 = 0.f;
        for (int cc = lane; cc < HDIM; cc += 32) {
            float x = lds[row * LNS + cc];
            p1 += x; p2 += x * x;
        }
        #pragma unroll
        for (int m = 16; m > 0; m >>= 1) {
            p1 += __shfl_xor(p1, m, 32);
            p2 += __shfl_xor(p2, m, 32);
        }
        const float mean = p1 * (1.0f / HDIM);
        const float var  = p2 * (1.0f / HDIM) - mean * mean;
        const float rstd = rsqrtf(var + EPS_LN);
        float* dst = acc + ((size_t)rowBase + row) * HDIM;
        for (int cc = lane; cc < HDIM; cc += 32) {
            float v = lnw[cc] * (lds[row * LNS + cc] - mean) * rstd + lnb[cc];
            if (beta) dst[cc] += v; else dst[cc] = v;
        }
    }
}

// ---------------------------------------------------------------------------
// Finalize: out = LN(acc + tte[token_type] + pos_enc), one block per token
// ---------------------------------------------------------------------------
__global__ __launch_bounds__(256)
void finalize_ln(const float* __restrict__ acc,
                 const float* __restrict__ tte,     // (T, HDIM)
                 const int*   __restrict__ ttids,   // (NTOK)
                 const float* __restrict__ lnw,
                 const float* __restrict__ lnb,
                 float* __restrict__ out)
{
    const int t   = blockIdx.x;
    const int tid = threadIdx.x;
    const int l   = t & 1023;                       // position within sequence
    const float* ar = acc + (size_t)t * HDIM;
    const float* tr = tte + (size_t)ttids[t] * HDIM;

    const float cpe = -9.210340371976184f / 768.0f; // -ln(10000)/H

    float xv[3];
    float p1 = 0.f, p2 = 0.f;
    #pragma unroll
    for (int k = 0; k < 3; ++k) {
        const int h  = tid + 256 * k;
        const int i2 = h & ~1;
        const float ang = (float)l * __expf((float)i2 * cpe);
        const float pe  = (h & 1) ? __cosf(ang) : __sinf(ang);
        float x = ar[h] + tr[h] + pe;
        xv[k] = x;
        p1 += x; p2 += x * x;
    }
    __shared__ float s1[256], s2[256];
    s1[tid] = p1; s2[tid] = p2;
    __syncthreads();
    for (int off = 128; off > 0; off >>= 1) {
        if (tid < off) { s1[tid] += s1[tid + off]; s2[tid] += s2[tid + off]; }
        __syncthreads();
    }
    const float mean = s1[0] * (1.0f / HDIM);
    const float var  = s2[0] * (1.0f / HDIM) - mean * mean;
    const float rstd = rsqrtf(var + EPS_LN);

    float* dst = out + (size_t)t * HDIM;
    #pragma unroll
    for (int k = 0; k < 3; ++k) {
        const int h = tid + 256 * k;
        dst[h] = lnw[h] * (xv[k] - mean) * rstd + lnb[h];
    }
}

// ---------------------------------------------------------------------------
// Launch
// ---------------------------------------------------------------------------
extern "C" void kernel_launch(void* const* d_in, const int* in_sizes, int n_in,
                              void* d_out, int out_size, void* d_ws, size_t ws_size,
                              hipStream_t stream)
{
    // inputs per setup_inputs() order
    const int*   input_ids   = (const int*)  d_in[0];
    const int*   input_ids2  = (const int*)  d_in[1];
    const float* video       = (const float*)d_in[2];
    const float* region      = (const float*)d_in[3];
    const int*   ttids       = (const int*)  d_in[4];
    const int*   ttids2      = (const int*)  d_in[5];
    const float* word_emb    = (const float*)d_in[6];
    const float* word_emb2   = (const float*)d_in[7];
    const float* tte         = (const float*)d_in[8];
    const float* tte2        = (const float*)d_in[9];

    const float* wfc_ln1_w  = (const float*)d_in[10];
    const float* wfc_ln1_b  = (const float*)d_in[11];
    const float* wfc_W      = (const float*)d_in[12];
    const float* wfc_b      = (const float*)d_in[13];
    const float* wfc_ln2_w  = (const float*)d_in[14];
    const float* wfc_ln2_b  = (const float*)d_in[15];

    const float* wfc2_ln1_w = (const float*)d_in[16];
    const float* wfc2_ln1_b = (const float*)d_in[17];
    const float* wfc2_W     = (const float*)d_in[18];
    const float* wfc2_b     = (const float*)d_in[19];
    const float* wfc2_ln2_w = (const float*)d_in[20];
    const float* wfc2_ln2_b = (const float*)d_in[21];

    const float* vid_ln1_w  = (const float*)d_in[22];
    const float* vid_ln1_b  = (const float*)d_in[23];
    const float* vid_W      = (const float*)d_in[24];
    const float* vid_b      = (const float*)d_in[25];
    const float* vid_ln2_w  = (const float*)d_in[26];
    const float* vid_ln2_b  = (const float*)d_in[27];

    const float* reg_ln1_w  = (const float*)d_in[28];
    const float* reg_ln1_b  = (const float*)d_in[29];
    const float* reg_W      = (const float*)d_in[30];
    const float* reg_b      = (const float*)d_in[31];
    const float* reg_ln2_w  = (const float*)d_in[32];
    const float* reg_ln2_b  = (const float*)d_in[33];

    const float* ln_w  = (const float*)d_in[34];
    const float* ln_b  = (const float*)d_in[35];
    const float* ln2_w = (const float*)d_in[36];
    const float* ln2_b = (const float*)d_in[37];
    (void)in_sizes; (void)n_in; (void)out_size; (void)ws_size;

    float* out = (float*)d_out;

    // workspace layout
    char* ws = (char*)d_ws;
    unsigned short* A   = (unsigned short*)ws;                          // 32768*3072*2 = 201326592 B
    unsigned short* Wb  = (unsigned short*)(ws + 201326592);            // 768*3072*2   = 4718592 B
    float*          acc = (float*)(ws + 201326592 + 4718592);           // 32768*768*4  = 100663296 B

    const dim3 tok(NTOK), thr(256), gtile(NTOK / 16);

    // ---------------- e1 = LN(word + video + tte + pe) ----------------
    ln1_to_bf16<<<tok, thr, 0, stream>>>(nullptr, input_ids, word_emb,
                                         wfc_ln1_w, wfc_ln1_b, A, 300, 320);
    conv_w_bf16<<<1024, thr, 0, stream>>>(wfc_W, Wb, 300, 320);
    gemm_bias_relu_ln<320><<<gtile, thr, 0, stream>>>(A, Wb, wfc_b,
                                         wfc_ln2_w, wfc_ln2_b, acc, 0);

    ln1_to_bf16<<<tok, thr, 0, stream>>>(video, nullptr, nullptr,
                                         vid_ln1_w, vid_ln1_b, A, 3072, 3072);
    conv_w_bf16<<<1024, thr, 0, stream>>>(vid_W, Wb, 3072, 3072);
    gemm_bias_relu_ln<3072><<<gtile, thr, 0, stream>>>(A, Wb, vid_b,
                                         vid_ln2_w, vid_ln2_b, acc, 1);

    finalize_ln<<<tok, thr, 0, stream>>>(acc, tte, ttids, ln_w, ln_b, out);

    // ---------------- e2 = LN(word2 + region + tte2 + pe) ----------------
    ln1_to_bf16<<<tok, thr, 0, stream>>>(nullptr, input_ids2, word_emb2,
                                         wfc2_ln1_w, wfc2_ln1_b, A, 300, 320);
    conv_w_bf16<<<1024, thr, 0, stream>>>(wfc2_W, Wb, 300, 320);
    gemm_bias_relu_ln<320><<<gtile, thr, 0, stream>>>(A, Wb, wfc2_b,
                                         wfc2_ln2_w, wfc2_ln2_b, acc, 0);

    ln1_to_bf16<<<tok, thr, 0, stream>>>(region, nullptr, nullptr,
                                         reg_ln1_w, reg_ln1_b, A, 2048, 2048);
    conv_w_bf16<<<1024, thr, 0, stream>>>(reg_W, Wb, 2048, 2048);
    gemm_bias_relu_ln<2048><<<gtile, thr, 0, stream>>>(A, Wb, reg_b,
                                         reg_ln2_w, reg_ln2_b, acc, 1);

    finalize_ln<<<tok, thr, 0, stream>>>(acc, tte2, ttids2, ln2_w, ln2_b,
                                         out + (size_t)NTOK * HDIM);
}
